// SenderGRULM_77927886618721
// MI455X (gfx1250) — matrix operand order repaired and checked by reading, not compiled, not run on hardware
//
#include <hip/hip_runtime.h>
#include <math.h>

// ---------------- problem constants ----------------
#define BB   512      // batch
#define EE   256      // embed dim
#define HH   512      // hidden
#define VV   4096     // vocab
#define F2   1024     // 2*F (concat prototypes)
#define G3H  1536     // 3*H
#define TT   18       // decode steps
#define ML   20       // total message length
#define SOS_IDX 1
#define EOS_IDX 2

typedef _Float16 v16h __attribute__((ext_vector_type(16)));
typedef _Float16 v8h  __attribute__((ext_vector_type(8)));
typedef float    v8f  __attribute__((ext_vector_type(8)));

// ---------------- WMMA fragment helpers (contiguous k order k = 16*(lane/16) + s, same for A and B) ----------------
// A (16x32, f16): lane l holds row m=l&15; lanes<16: K=0..15, lanes>=16: K=16..31
__device__ inline v16h loadA(const _Float16* __restrict__ X, int ldk, int m_base, int k_base, int lane) {
    const _Float16* p = X + (size_t)(m_base + (lane & 15)) * ldk + k_base + ((lane & 16) ? 16 : 0);
    return *(const v16h*)(p);
}
// B (32x16, f16) with B[k,n] = W[n_global, k]: lane l holds column n=l&15 (a row of W);
// lanes<16: K=k_base+0..15, lanes>=16: K=k_base+16..31 (one contiguous 32B read)
__device__ inline v16h loadB(const _Float16* __restrict__ W, int ldk, int n_base, int k_base, int lane) {
    const _Float16* p = W + (size_t)(n_base + (lane & 15)) * ldk + k_base + ((lane & 16) ? 16 : 0);
    return *(const v16h*)(p);
}
__device__ inline v8f wmma32(v16h a, v16h b, v8f c) {
    return __builtin_amdgcn_wmma_f32_16x16x32_f16(false, a, false, b, (short)0, c, false, false);
}

// Double-buffered multi-tile GEMM body: one wave accumulates NB 16x16 tiles
// (columns n0 + j*NSTRIDE) over a K-loop, prefetching the next K-slice's
// fragments before consuming the current ones so loads overlap the WMMAs.
template <int NB, int NSTRIDE>
__device__ inline void gemm_nb(const _Float16* __restrict__ A, int lda, int m_base,
                               const _Float16* __restrict__ W, int ldw, int n0,
                               int K, int lane, v8f acc[NB]) {
    v16h aC = loadA(A, lda, m_base, 0, lane);
    v16h bC[NB];
#pragma unroll
    for (int j = 0; j < NB; ++j) bC[j] = loadB(W, ldw, n0 + j * NSTRIDE, 0, lane);
    for (int k = 32; k < K; k += 32) {
        v16h aN = loadA(A, lda, m_base, k, lane);          // prefetch next slice
        v16h bN[NB];
#pragma unroll
        for (int j = 0; j < NB; ++j) bN[j] = loadB(W, ldw, n0 + j * NSTRIDE, k, lane);
#pragma unroll
        for (int j = 0; j < NB; ++j) acc[j] = wmma32(aC, bC[j], acc[j]);   // consume current
        aC = aN;
#pragma unroll
        for (int j = 0; j < NB; ++j) bC[j] = bN[j];
    }
#pragma unroll
    for (int j = 0; j < NB; ++j) acc[j] = wmma32(aC, bC[j], acc[j]);       // epilogue
}

__device__ inline float fsigmoid(float x) { return 1.0f / (1.0f + __expf(-x)); }
__device__ inline float ftanh(float x) {
    float ax = fabsf(x);
    float e  = __expf(2.0f * ax);
    float t  = 1.0f - 2.0f / (e + 1.0f);
    return copysignf(t, x);
}

// ---------------- init / conversion kernels ----------------
__global__ void k_convert(const float* __restrict__ in, _Float16* __restrict__ out, int n) {
    for (int i = blockIdx.x * blockDim.x + threadIdx.x; i < n; i += gridDim.x * blockDim.x)
        out[i] = (_Float16)in[i];
}

// Xinit[b, 0:512]=proto_a[b,:], Xinit[b, 512:1024]=proto_b[b,:] (f16)
__global__ void k_cat(const float* __restrict__ pa, const float* __restrict__ pb,
                      _Float16* __restrict__ Xi) {
    int i = blockIdx.x * blockDim.x + threadIdx.x;       // over BB*F2, launched exactly
    int b = i >> 10, c = i & 1023;
    float v = (c < 512) ? pa[b * 512 + c] : pb[b * 512 + (c - 512)];
    Xi[i] = (_Float16)v;
}

// x0 = emb[SOS_IDX] broadcast to every batch row (f16)
__global__ void k_x0(const float* __restrict__ emb, _Float16* __restrict__ xb) {
    int i = blockIdx.x * blockDim.x + threadIdx.x;       // over BB*EE
    int e = i & (EE - 1);
    xb[i] = (_Float16)emb[SOS_IDX * EE + e];
}

__global__ void k_zero(float4* __restrict__ out, int n4) {
    float4 z = make_float4(0.f, 0.f, 0.f, 0.f);
    for (int i = blockIdx.x * blockDim.x + threadIdx.x; i < n4; i += gridDim.x * blockDim.x)
        out[i] = z;
}

__global__ void k_soseos(float* __restrict__ lang) {
    int b = blockIdx.x * blockDim.x + threadIdx.x;       // over BB
    lang[((size_t)b * ML + 0)      * VV + SOS_IDX] = 1.0f;
    lang[((size_t)b * ML + ML - 1) * VV + EOS_IDX] = 1.0f;
}

// ---------------- h0 = Xinit @ W_init^T + b_init  (512x512x1024) ----------------
// 256 waves; each wave computes a 16x64 output slab (4 WMMA tiles, A-frag reuse)
__global__ __launch_bounds__(256) void k_h0(const _Float16* __restrict__ Xi,
                                            const _Float16* __restrict__ Wi,
                                            const float* __restrict__ b_init,
                                            float* __restrict__ h32, _Float16* __restrict__ hb) {
    int w = (blockIdx.x * 256 + threadIdx.x) >> 5;
    int lane = threadIdx.x & 31;
    int m_base = (w >> 3) * 16;       // 32 row tiles
    int n_base = (w & 7) * 64;        // 8 column slabs of 64
    v8f acc[4] = {};
    gemm_nb<4, 16>(Xi, F2, m_base, Wi, F2, n_base, F2, lane, acc);
    int n = lane & 15, mo = (lane >> 4) * 8;
#pragma unroll
    for (int j = 0; j < 4; ++j) {
        int col = n_base + j * 16 + n;
        float bias = b_init[col];
#pragma unroll
        for (int i = 0; i < 8; ++i) {
            int row = m_base + mo + i;
            float v = acc[j][i] + bias;
            h32[(size_t)row * HH + col] = v;
            hb [(size_t)row * HH + col] = (_Float16)v;
        }
    }
}

// ---------------- GRU cell: 6 simultaneous WMMA accumulations per wave ----------------
// 1024 waves; wave tile = (16 batch rows) x (16 hidden cols), r/z/n gates together
__global__ __launch_bounds__(256) void k_gru(const _Float16* __restrict__ xb,
                                             const _Float16* __restrict__ hb,
                                             const float* __restrict__ h32,
                                             const _Float16* __restrict__ Wih,
                                             const _Float16* __restrict__ Whh,
                                             const float* __restrict__ b_ih,
                                             const float* __restrict__ b_hh,
                                             float* __restrict__ h32n, _Float16* __restrict__ hbn) {
    int w = (blockIdx.x * 256 + threadIdx.x) >> 5;
    int lane = threadIdx.x & 31;
    int m_base = (w >> 5) * 16;       // batch tile
    int i_base = (w & 31) * 16;       // hidden tile
    v8f gi[3] = {};                   // ir, iz, in
    v8f gh[3] = {};                   // hr, hz, hn
    gemm_nb<3, HH>(xb, EE, m_base, Wih, EE, i_base, EE, lane, gi);   // x @ W_ih^T
    gemm_nb<3, HH>(hb, HH, m_base, Whh, HH, i_base, HH, lane, gh);   // h @ W_hh^T
    int n = lane & 15, mo = (lane >> 4) * 8;
    int col = i_base + n;
    float bir = b_ih[col], biz = b_ih[col + HH], bin = b_ih[col + 2 * HH];
    float bhr = b_hh[col], bhz = b_hh[col + HH], bhn = b_hh[col + 2 * HH];
#pragma unroll
    for (int i = 0; i < 8; ++i) {
        int row = m_base + mo + i;
        float r  = fsigmoid((gi[0][i] + bir) + (gh[0][i] + bhr));
        float z  = fsigmoid((gi[1][i] + biz) + (gh[1][i] + bhz));
        float ng = ftanh((gi[2][i] + bin) + r * (gh[2][i] + bhn));
        float ho = h32[(size_t)row * HH + col];
        float v  = (1.0f - z) * ng + z * ho;
        h32n[(size_t)row * HH + col] = v;
        hbn [(size_t)row * HH + col] = (_Float16)v;
    }
}

// ---------------- logits = h_new @ W_out^T + b_out + Gumbel(u) ----------------
// 2048 waves; wave = 16x64 slab; scores land in an 8MB L2-resident scratch
__global__ __launch_bounds__(256) void k_logits(const _Float16* __restrict__ hbn,
                                                const _Float16* __restrict__ Wout,
                                                const float* __restrict__ b_out,
                                                const float* __restrict__ ug, int t,
                                                float* __restrict__ scores) {
    int w = (blockIdx.x * 256 + threadIdx.x) >> 5;
    int lane = threadIdx.x & 31;
    int m_base = (w >> 6) * 16;       // 32 batch tiles
    int n_base = (w & 63) * 64;       // 64 vocab slabs of 64
    v8f acc[4] = {};
    gemm_nb<4, 16>(hbn, HH, m_base, Wout, HH, n_base, HH, lane, acc);
    int n = lane & 15, mo = (lane >> 4) * 8;
    const float* u = ug + (size_t)t * BB * VV;
#pragma unroll
    for (int j = 0; j < 4; ++j) {
        int col = n_base + j * 16 + n;
        float bias = b_out[col];
#pragma unroll
        for (int i = 0; i < 8; ++i) {
            int row = m_base + mo + i;
            float uu = u[(size_t)row * VV + col];
            float g  = -__logf(-__logf(uu));        // Gumbel(0,1); TAU == 1
            scores[(size_t)row * VV + col] = acc[j][i] + bias + g;
        }
    }
}

// ---------------- sample: per-row argmax + softmax value, emit one-hot & next x ----------------
__global__ __launch_bounds__(256) void k_sample(const float* __restrict__ scores,
                                                const float* __restrict__ emb, int t,
                                                float* __restrict__ lang,
                                                _Float16* __restrict__ xb) {
    __shared__ float sm[256];
    __shared__ int   si[256];
    __shared__ float red[256];
    __shared__ float s_val;
    __shared__ int   s_tok;
    int b = blockIdx.x, tid = threadIdx.x;
    const float* srow = scores + (size_t)b * VV;

    float mx = -INFINITY; int ax = 0;
    for (int v = tid; v < VV; v += 256) {                 // first-occurrence argmax
        float s = srow[v];
        if (s > mx) { mx = s; ax = v; }
    }
    sm[tid] = mx; si[tid] = ax;
    __syncthreads();
    for (int off = 128; off > 0; off >>= 1) {
        if (tid < off) {
            float m2 = sm[tid + off]; int i2 = si[tid + off];
            if (m2 > sm[tid] || (m2 == sm[tid] && i2 < si[tid])) { sm[tid] = m2; si[tid] = i2; }
        }
        __syncthreads();
    }
    float fmx = sm[0]; int tok = si[0];

    float se = 0.0f;
    for (int v = tid; v < VV; v += 256) se += __expf(srow[v] - fmx);
    red[tid] = se;
    __syncthreads();
    for (int off = 128; off > 0; off >>= 1) {
        if (tid < off) red[tid] += red[tid + off];
        __syncthreads();
    }
    if (tid == 0) {
        float y   = 1.0f / red[0];                        // softmax prob at the argmax
        float val = (1.0f - y) + y;                       // straight-through: hard - y + y
        lang[((size_t)b * ML + (t + 1)) * VV + tok] = val;
        s_val = val; s_tok = tok;
    }
    __syncthreads();
    // x_next = val * emb[tok]  (blockDim == EE == 256)
    xb[(size_t)b * EE + tid] = (_Float16)(s_val * emb[(size_t)s_tok * EE + tid]);
}

// ---------------- host side ----------------
extern "C" void kernel_launch(void* const* d_in, const int* in_sizes, int n_in,
                              void* d_out, int out_size, void* d_ws, size_t ws_size,
                              hipStream_t stream) {
    const float* proto_a = (const float*)d_in[0];
    const float* proto_b = (const float*)d_in[1];
    const float* W_init  = (const float*)d_in[2];
    const float* b_init  = (const float*)d_in[3];
    const float* W_ih    = (const float*)d_in[4];
    const float* W_hh    = (const float*)d_in[5];
    const float* b_ih    = (const float*)d_in[6];
    const float* b_hh    = (const float*)d_in[7];
    const float* W_out   = (const float*)d_in[8];
    const float* b_out   = (const float*)d_in[9];
    const float* emb     = (const float*)d_in[10];
    const float* ug      = (const float*)d_in[11];
    float* lang = (float*)d_out;

    // workspace carve-out (256B aligned)
    char* p = (char*)d_ws;
    auto alloc = [&](size_t bytes) -> char* {
        char* r = p; p += (bytes + 255) & ~(size_t)255; return r;
    };
    _Float16* Wih_h   = (_Float16*)alloc((size_t)G3H * EE * 2);
    _Float16* Whh_h   = (_Float16*)alloc((size_t)G3H * HH * 2);
    _Float16* Wout_h  = (_Float16*)alloc((size_t)VV  * HH * 2);
    _Float16* Winit_h = (_Float16*)alloc((size_t)HH  * F2 * 2);
    _Float16* Xinit_h = (_Float16*)alloc((size_t)BB  * F2 * 2);
    _Float16* xb      = (_Float16*)alloc((size_t)BB  * EE * 2);
    float*    h32a    = (float*)   alloc((size_t)BB  * HH * 4);
    float*    h32b    = (float*)   alloc((size_t)BB  * HH * 4);
    _Float16* hba     = (_Float16*)alloc((size_t)BB  * HH * 2);
    _Float16* hbb     = (_Float16*)alloc((size_t)BB  * HH * 2);
    float*    scores  = (float*)   alloc((size_t)BB  * VV * 4);

    // one-time per call: f16 weight conversion (L2-resident afterwards)
    k_convert<<<1536, 256, 0, stream>>>(W_ih,   Wih_h,   G3H * EE);
    k_convert<<<3072, 256, 0, stream>>>(W_hh,   Whh_h,   G3H * HH);
    k_convert<<<8192, 256, 0, stream>>>(W_out,  Wout_h,  VV  * HH);
    k_convert<<<2048, 256, 0, stream>>>(W_init, Winit_h, HH  * F2);
    k_cat<<<(BB * F2) / 256, 256, 0, stream>>>(proto_a, proto_b, Xinit_h);
    k_x0 <<<(BB * EE) / 256, 256, 0, stream>>>(emb, xb);

    // output: zero then SOS/EOS one-hots
    int n4 = out_size / 4;
    k_zero  <<<4096, 256, 0, stream>>>((float4*)lang, n4);
    k_soseos<<<BB / 256, 256, 0, stream>>>(lang);

    // h0
    k_h0<<<32, 256, 0, stream>>>(Xinit_h, Winit_h, b_init, h32a, hba);

    // 18 sequential decode steps (ping-pong h buffers)
    for (int t = 0; t < TT; ++t) {
        float*    hc  = (t & 1) ? h32b : h32a;
        float*    hn_ = (t & 1) ? h32a : h32b;
        _Float16* hbc = (t & 1) ? hbb  : hba;
        _Float16* hbn = (t & 1) ? hba  : hbb;
        k_gru   <<<128, 256, 0, stream>>>(xb, hbc, hc, Wih_h, Whh_h, b_ih, b_hh, hn_, hbn);
        k_logits<<<256, 256, 0, stream>>>(hbn, Wout_h, b_out, ug, t, scores);
        k_sample<<<BB,  256, 0, stream>>>(scores, emb, t, lang, xb);
    }
}
